// CrossAttentionFusion_32607391711727
// MI455X (gfx1250) — matrix operand order repaired and checked by reading, not compiled, not run on hardware
//
#include <hip/hip_runtime.h>

typedef __attribute__((ext_vector_type(16))) __bf16 bf16x16;
typedef __attribute__((ext_vector_type(2)))  __bf16 bf16x2;
typedef __attribute__((ext_vector_type(8)))  float  f32x8;

union BF16Frag { bf16x16 v; unsigned short u[16]; };

__device__ __forceinline__ unsigned short f2bf(float f) {
    unsigned int u = __float_as_uint(f);
    unsigned int r = (u + 0x7FFFu + ((u >> 16) & 1u)) >> 16;
    return (unsigned short)r;
}
__device__ __forceinline__ unsigned int pk2bf(float a, float b) {
#if __has_builtin(__builtin_amdgcn_cvt_pk_bf16_f32)
    bf16x2 t = __builtin_amdgcn_cvt_pk_bf16_f32(a, b);
    union { bf16x2 h; unsigned int w; } cv;
    cv.h = t;
    return cv.w;
#else
    return (unsigned int)f2bf(a) | ((unsigned int)f2bf(b) << 16);
#endif
}

// A-fragment (16x32, 16-bit) contraction index for slot j, lane-half ah.
// ISA 7.12.2: lanes 0-15: V0..V3 = K 0..7, V4..V7 = K 16..23; lanes 16-31: +8.
__device__ __forceinline__ int a_kidx(int j, int ah) {
    int v = j >> 1, hi = j & 1;
    return (v < 4) ? (ah * 8 + 2 * v + hi) : (16 + ah * 8 + 2 * (v - 4) + hi);
}

__device__ __forceinline__ float half_reduce_max(float x) {
    #pragma unroll
    for (int m = 1; m < 16; m <<= 1) x = fmaxf(x, __shfl_xor(x, m, 32));
    return x;
}
__device__ __forceinline__ float half_reduce_sum(float x) {
    #pragma unroll
    for (int m = 1; m < 16; m <<= 1) x += __shfl_xor(x, m, 32);
    return x;
}

// ---------------------------------------------------------------------------
// GEMM: C[M,N] = A[M,K] @ W[K,N] + bias.  Block tile 128(M) x 64(N), K-step 32.
// 128 threads = 4 waves; each wave computes a 32x64 slab (8 WMMAs / K-step).
// Software-pipelined: tile k+1 is loaded into registers while WMMAs consume
// tile k from LDS; registers are committed to LDS at the top of the next step.
// MODE 0: scatter bf16 -> dst0 head-major (B,H,S,HD)   (Q projection)
// MODE 1: cols <1024 -> dst0 (K), >=1024 -> dst1 (V)   (KV projection)
// MODE 2: f32 row-major -> dstF                        (output projection)
// ABF: A operand is bf16 (true) or f32 (false).
// ---------------------------------------------------------------------------
template <bool ABF, int MODE>
__global__ __launch_bounds__(128) void gemm_wmma_bf16(
    const void* __restrict__ Aptr,
    const float* __restrict__ W,
    const float* __restrict__ bias,
    int M, int N, int K,
    unsigned short* __restrict__ dst0,
    unsigned short* __restrict__ dst1,
    float* __restrict__ dstF)
{
    __shared__ unsigned short As[128][32];  // [m][k] bf16 (8 KB)
    __shared__ unsigned short WsT[64][32];  // [n][k] bf16 transposed (4 KB)

    const int tid  = threadIdx.x;
    const int lane = tid & 31;
    const int wv   = tid >> 5;
    const int ah   = lane >> 4;
    const int n16  = lane & 15;
    const int m0   = blockIdx.y * 128;
    const int n0   = blockIdx.x * 64;

    f32x8 acc[2][4];
    #pragma unroll
    for (int sf = 0; sf < 2; ++sf)
        #pragma unroll
        for (int t = 0; t < 4; ++t)
            #pragma unroll
            for (int r = 0; r < 8; ++r) acc[sf][t][r] = 0.f;

    const float*          Af = (const float*)Aptr;
    const unsigned short* Ab = (const unsigned short*)Aptr;

    // Per-thread staging registers for one (A, W) tile pair.
    float4 arf[8];
    uint2  arb[8];
    float  wr[16];

    const int ar_r  = tid >> 3, ar_c4 = (tid & 7) * 4;     // A: 1024 float4 groups
    const int ws_c  = tid & 63, ws_r0 = (tid >> 6) * 4;    // W: 512 4-row segments

    #define GEMM_LOAD_TILES(K0)                                                   \
        do {                                                                      \
            _Pragma("unroll")                                                     \
            for (int i = 0; i < 8; ++i) {                                         \
                int r = ar_r + i * 16;                                            \
                if (ABF) arb[i] = *(const uint2*)(Ab + (size_t)(m0 + r) * K + (K0) + ar_c4); \
                else     arf[i] = *(const float4*)(Af + (size_t)(m0 + r) * K + (K0) + ar_c4); \
            }                                                                     \
            _Pragma("unroll")                                                     \
            for (int i = 0; i < 4; ++i) {                                         \
                int r0 = ws_r0 + i * 8;                                           \
                const float* wp = W + (size_t)((K0) + r0) * N + (n0 + ws_c);      \
                wr[4 * i + 0] = wp[0];                                            \
                wr[4 * i + 1] = wp[(size_t)N];                                    \
                wr[4 * i + 2] = wp[(size_t)2 * N];                                \
                wr[4 * i + 3] = wp[(size_t)3 * N];                                \
            }                                                                     \
        } while (0)

    GEMM_LOAD_TILES(0);

    for (int k0 = 0; k0 < K; k0 += 32) {
        __syncthreads();
        // ---- commit staged registers to LDS ----
        #pragma unroll
        for (int i = 0; i < 8; ++i) {
            unsigned int* d = (unsigned int*)&As[ar_r + i * 16][ar_c4];
            if (ABF) { d[0] = arb[i].x; d[1] = arb[i].y; }
            else     { d[0] = pk2bf(arf[i].x, arf[i].y); d[1] = pk2bf(arf[i].z, arf[i].w); }
        }
        #pragma unroll
        for (int i = 0; i < 4; ++i) {
            unsigned int* d = (unsigned int*)&WsT[ws_c][ws_r0 + i * 8];
            d[0] = pk2bf(wr[4 * i + 0], wr[4 * i + 1]);
            d[1] = pk2bf(wr[4 * i + 2], wr[4 * i + 3]);
        }
        __syncthreads();

        // ---- kick off next tile's global loads (overlaps with WMMA) ----
        if (k0 + 32 < K) {
            GEMM_LOAD_TILES(k0 + 32);
        }
        if (k0 + 64 < K) {   // long-range prefetch into cache hierarchy
            const size_t esz = ABF ? 2 : 4;
            __builtin_prefetch((const char*)Aptr + ((size_t)(m0 + ar_r) * K + k0 + 64) * esz, 0, 0);
            __builtin_prefetch((const char*)(W + (size_t)(k0 + 64 + ws_r0) * N + n0 + ws_c), 0, 0);
        }

        // ---- WMMA on the tile now in LDS ----
        BF16Frag a[2];
        #pragma unroll
        for (int sf = 0; sf < 2; ++sf)
            #pragma unroll
            for (int j = 0; j < 16; ++j)
                a[sf].u[j] = As[wv * 32 + sf * 16 + n16][a_kidx(j, ah)];

        #pragma unroll
        for (int t = 0; t < 4; ++t) {
            BF16Frag b;
            #pragma unroll
            for (int j = 0; j < 16; ++j) b.u[j] = WsT[t * 16 + n16][a_kidx(j, ah)];
            #pragma unroll
            for (int sf = 0; sf < 2; ++sf)
                acc[sf][t] = __builtin_amdgcn_wmma_f32_16x16x32_bf16(
                    false, a[sf].v, false, b.v, (short)0, acc[sf][t], false, false);
        }
    }
    #undef GEMM_LOAD_TILES

    // Epilogue: bias + scatter. C/D layout: row = half*8 + r, col = lane&15.
    #pragma unroll
    for (int t = 0; t < 4; ++t) {
        int gc = n0 + t * 16 + n16;
        float bv = bias[gc];
        #pragma unroll
        for (int sf = 0; sf < 2; ++sf)
            #pragma unroll
            for (int r = 0; r < 8; ++r) {
                int gm = m0 + wv * 32 + sf * 16 + ah * 8 + r;
                float val = acc[sf][t][r] + bv;
                if (MODE == 2) {
                    dstF[(size_t)gm * N + gc] = val;
                } else {
                    int bb = gm >> 10, ss = gm & 1023;   // S = 1024
                    int col = gc;
                    unsigned short* d = dst0;
                    if (MODE == 1 && col >= 1024) { col -= 1024; d = dst1; }
                    int h = col >> 6, dd = col & 63;     // HD = 64
                    d[(((size_t)(bb * 16 + h) << 10) + ss) * 64 + dd] = f2bf(val);
                }
            }
    }
}

// ---------------------------------------------------------------------------
// Flash attention for one (b, h, 64-query tile). 128 threads = 4 waves,
// each wave owns 16 queries. Keys processed in tiles of 64, software-pipelined.
// ---------------------------------------------------------------------------
template <bool CLAMP>
__global__ __launch_bounds__(128) void attn_wmma(
    const unsigned short* __restrict__ Qb,
    const unsigned short* __restrict__ Kb,
    const unsigned short* __restrict__ Vb,
    unsigned short* __restrict__ fused,
    int dir)
{
    __shared__ unsigned short Ks[64][64];      // [key][d]
    __shared__ unsigned short VT[64][64];      // [d][key]
    __shared__ unsigned short Ps[4][16][64];   // per-wave P tile [q][key]

    const int tid  = threadIdx.x;
    const int lane = tid & 31;
    const int wv   = tid >> 5;
    const int ah   = lane >> 4;
    const int n16  = lane & 15;
    const int b = blockIdx.z, h = blockIdx.y, qt = blockIdx.x;
    const size_t bh_off = (size_t)(b * 16 + h) * 1024 * 64;
    const int qbase = qt * 64 + wv * 16;
    const float scale = 0.125f;                // HD^-0.5

    BF16Frag aQ[2];
    {
        const unsigned short* Qrow = Qb + bh_off + (size_t)(qbase + n16) * 64;
        #pragma unroll
        for (int c = 0; c < 2; ++c)
            #pragma unroll
            for (int j = 0; j < 16; ++j) aQ[c].u[j] = Qrow[c * 32 + a_kidx(j, ah)];
    }

    float m_i[8], l_i[8];
    f32x8 O[4];
    #pragma unroll
    for (int r = 0; r < 8; ++r) { m_i[r] = -1e30f; l_i[r] = 0.f; }
    #pragma unroll
    for (int t = 0; t < 4; ++t)
        #pragma unroll
        for (int r = 0; r < 8; ++r) O[t][r] = 0.f;

    // Per-thread staging registers for one (K, V^T) tile pair.
    uint4        krg[4];
    unsigned int vrg[8][2];
    const int kr_r = tid >> 3, kr_c8 = (tid & 7) * 8;     // K: 512 groups of 8
    const int vs_c = tid & 63, vs_r0 = (tid >> 6) * 4;    // V: 1024 4-key segments

    #define ATTN_LOAD_TILES(KB)                                                    \
        do {                                                                       \
            _Pragma("unroll")                                                      \
            for (int i = 0; i < 4; ++i)                                            \
                krg[i] = *(const uint4*)(Kb + bh_off + (size_t)((KB) + kr_r + i * 16) * 64 + kr_c8); \
            _Pragma("unroll")                                                      \
            for (int i = 0; i < 8; ++i) {                                          \
                const unsigned short* vp =                                         \
                    Vb + bh_off + (size_t)((KB) + vs_r0 + i * 8) * 64 + vs_c;      \
                vrg[i][0] = (unsigned int)vp[0]   | ((unsigned int)vp[64]  << 16); \
                vrg[i][1] = (unsigned int)vp[128] | ((unsigned int)vp[192] << 16); \
            }                                                                      \
        } while (0)

    ATTN_LOAD_TILES(0);

    for (int kb = 0; kb < 1024; kb += 64) {
        __syncthreads();                       // all waves done reading prev tile
        // ---- commit staged registers to LDS ----
        #pragma unroll
        for (int i = 0; i < 4; ++i)
            *(uint4*)&Ks[kr_r + i * 16][kr_c8] = krg[i];
        #pragma unroll
        for (int i = 0; i < 8; ++i) {
            unsigned int* d = (unsigned int*)&VT[vs_c][vs_r0 + i * 8];
            d[0] = vrg[i][0]; d[1] = vrg[i][1];
        }
        __syncthreads();

        // ---- kick off next tile's global loads (overlaps with compute) ----
        if (kb + 64 < 1024) {
            ATTN_LOAD_TILES(kb + 64);
        }
        if (kb + 128 < 1024) {
            __builtin_prefetch((const char*)(Kb + bh_off + (size_t)(kb + 128 + kr_r) * 64 + kr_c8), 0, 0);
            __builtin_prefetch((const char*)(Vb + bh_off + (size_t)(kb + 128 + vs_r0) * 64 + vs_c), 0, 0);
        }

        // scores: 16q x 64k via WMMA (contraction over HD=64 -> 2 chained)
        f32x8 sc[4];
        #pragma unroll
        for (int t = 0; t < 4; ++t) {
            #pragma unroll
            for (int r = 0; r < 8; ++r) sc[t][r] = 0.f;
            #pragma unroll
            for (int c = 0; c < 2; ++c) {
                BF16Frag bK;
                #pragma unroll
                for (int j = 0; j < 16; ++j)
                    bK.u[j] = Ks[t * 16 + n16][c * 32 + a_kidx(j, ah)];
                sc[t] = __builtin_amdgcn_wmma_f32_16x16x32_bf16(
                    false, aQ[c].v, false, bK.v, (short)0, sc[t], false, false);
            }
        }

        // online softmax
        #pragma unroll
        for (int r = 0; r < 8; ++r) {
            float mx = -1e30f;
            #pragma unroll
            for (int t = 0; t < 4; ++t) {
                float s = sc[t][r] * scale;
                if (CLAMP) s = fminf(fmaxf(s, -100.f), 100.f);
                sc[t][r] = s;
                mx = fmaxf(mx, s);
            }
            mx = half_reduce_max(mx);
            float mnew  = fmaxf(m_i[r], mx);
            float alpha = __expf(m_i[r] - mnew);
            m_i[r] = mnew;
            float rsum = 0.f;
            #pragma unroll
            for (int t = 0; t < 4; ++t) {
                float p = __expf(sc[t][r] - mnew);
                rsum += p;
                Ps[wv][ah * 8 + r][t * 16 + n16] = f2bf(p);
            }
            rsum = half_reduce_sum(rsum);
            l_i[r] = l_i[r] * alpha + rsum;
            #pragma unroll
            for (int t = 0; t < 4; ++t) O[t][r] *= alpha;
        }
        __syncthreads();

        // O += P @ V  (contraction over 64 keys -> 2 chained WMMAs)
        #pragma unroll
        for (int c = 0; c < 2; ++c) {
            BF16Frag aP;
            #pragma unroll
            for (int j = 0; j < 16; ++j)
                aP.u[j] = Ps[wv][n16][c * 32 + a_kidx(j, ah)];
            #pragma unroll
            for (int t = 0; t < 4; ++t) {
                BF16Frag bV;
                #pragma unroll
                for (int j = 0; j < 16; ++j)
                    bV.u[j] = VT[t * 16 + n16][c * 32 + a_kidx(j, ah)];
                O[t] = __builtin_amdgcn_wmma_f32_16x16x32_bf16(
                    false, aP.v, false, bV.v, (short)0, O[t], false, false);
            }
        }
    }
    #undef ATTN_LOAD_TILES

    // normalize + write into concatenated fused buffer (B,S,2048) bf16
    #pragma unroll
    for (int t = 0; t < 4; ++t)
        #pragma unroll
        for (int r = 0; r < 8; ++r) {
            int q = qbase + ah * 8 + r;
            int d = t * 16 + n16;
            float v = O[t][r] / l_i[r];
            fused[(size_t)(b * 1024 + q) * 2048 + dir * 1024 + h * 64 + d] = f2bf(v);
        }
}

// ---------------------------------------------------------------------------
// LayerNorm over last dim (1024) — one row per block, float4 I/O.
// ---------------------------------------------------------------------------
__global__ __launch_bounds__(256) void layernorm_kernel(
    const float* __restrict__ x, const float* __restrict__ gamma,
    const float* __restrict__ beta, float* __restrict__ out)
{
    __shared__ float sh0[256], sh1[256];
    int row = blockIdx.x;
    const float4 xv = *(const float4*)(x + (size_t)row * 1024 + threadIdx.x * 4);
    float s  = xv.x + xv.y + xv.z + xv.w;
    float s2 = xv.x * xv.x + xv.y * xv.y + xv.z * xv.z + xv.w * xv.w;
    sh0[threadIdx.x] = s; sh1[threadIdx.x] = s2;
    __syncthreads();
    for (int o = 128; o > 0; o >>= 1) {
        if (threadIdx.x < o) {
            sh0[threadIdx.x] += sh0[threadIdx.x + o];
            sh1[threadIdx.x] += sh1[threadIdx.x + o];
        }
        __syncthreads();
    }
    float mu  = sh0[0] * (1.f / 1024.f);
    float var = sh1[0] * (1.f / 1024.f) - mu * mu;
    float inv = rsqrtf(var + 1e-5f);
    const float4 gv = *(const float4*)(gamma + threadIdx.x * 4);
    const float4 bv = *(const float4*)(beta + threadIdx.x * 4);
    float4 ov;
    ov.x = (xv.x - mu) * inv * gv.x + bv.x;
    ov.y = (xv.y - mu) * inv * gv.y + bv.y;
    ov.z = (xv.z - mu) * inv * gv.z + bv.z;
    ov.w = (xv.w - mu) * inv * gv.w + bv.w;
    *(float4*)(out + (size_t)row * 1024 + threadIdx.x * 4) = ov;
}

extern "C" void kernel_launch(void* const* d_in, const int* in_sizes, int n_in,
                              void* d_out, int out_size, void* d_ws, size_t ws_size,
                              hipStream_t stream) {
    (void)in_sizes; (void)n_in; (void)out_size; (void)ws_size;
    const float* img   = (const float*)d_in[0];
    const float* wav   = (const float*)d_in[1];
    const float* Wq_i  = (const float*)d_in[2];
    const float* bq_i  = (const float*)d_in[3];
    const float* Wkv_w = (const float*)d_in[4];
    const float* bkv_w = (const float*)d_in[5];
    const float* Wq_w  = (const float*)d_in[6];
    const float* bq_w  = (const float*)d_in[7];
    const float* Wkv_i = (const float*)d_in[8];
    const float* bkv_i = (const float*)d_in[9];
    const float* Wp    = (const float*)d_in[10];
    const float* bp    = (const float*)d_in[11];
    const float* gamma = (const float*)d_in[12];
    const float* beta  = (const float*)d_in[13];
    float* out = (float*)d_out;

    char* ws = (char*)d_ws;
    const size_t MB = (size_t)1 << 20;
    unsigned short* fused = (unsigned short*)(ws);            // 16 MB bf16 (4096x2048)
    unsigned short* Qb    = (unsigned short*)(ws + 16 * MB);  // 8 MB
    unsigned short* Kb    = (unsigned short*)(ws + 24 * MB);  // 8 MB
    unsigned short* Vb    = (unsigned short*)(ws + 32 * MB);  // 8 MB
    float*          tmp   = (float*)(ws + 16 * MB);           // aliases Qb/Kb after attn

    dim3 blk(128);
    // direction 0: image queries -> waveform K/V (clamped scores)
    gemm_wmma_bf16<false, 0><<<dim3(16, 32), blk, 0, stream>>>(img, Wq_i, bq_i, 4096, 1024, 1024, Qb, nullptr, nullptr);
    gemm_wmma_bf16<false, 1><<<dim3(32, 32), blk, 0, stream>>>(wav, Wkv_w, bkv_w, 4096, 2048, 1024, Kb, Vb, nullptr);
    attn_wmma<true><<<dim3(16, 16, 4), blk, 0, stream>>>(Qb, Kb, Vb, fused, 0);
    // direction 1: waveform queries -> image K/V (no clamp)
    gemm_wmma_bf16<false, 0><<<dim3(16, 32), blk, 0, stream>>>(wav, Wq_w, bq_w, 4096, 1024, 1024, Qb, nullptr, nullptr);
    gemm_wmma_bf16<false, 1><<<dim3(32, 32), blk, 0, stream>>>(img, Wkv_i, bkv_i, 4096, 2048, 1024, Kb, Vb, nullptr);
    attn_wmma<false><<<dim3(16, 16, 4), blk, 0, stream>>>(Qb, Kb, Vb, fused, 1);
    // output projection + LayerNorm
    gemm_wmma_bf16<true, 2><<<dim3(16, 32), blk, 0, stream>>>(fused, Wp, bp, 4096, 1024, 2048, nullptr, nullptr, tmp);
    layernorm_kernel<<<4096, 256, 0, stream>>>(tmp, gamma, beta, out);
}
